// DeepseekV2Attention_44616120271588
// MI455X (gfx1250) — matrix operand order repaired and checked by reading, not compiled, not run on hardware
//
#include <hip/hip_runtime.h>
#include <hip/hip_bf16.h>
#include <math.h>
#include <stdint.h>

// ---------------------------------------------------------------------------
// DeepSeek-V2 MLA prefill for MI455X (gfx1250, wave32, WMMA).
//  - all matmuls on v_wmma_f32_16x16x32_f16 (128x128 block tiles, 8 waves)
//  - grid.x = M tiles so one L2-resident B stripe serves 8 consecutive blocks
//  - q_full/k_full/attn kept in f16 (SCALE folded into q_full)
//  - flash staging via global_load_async_to_lds_b128 (+ s_wait_asynccnt)
//  - flash attention: 16-query tile x head per block, 64-key tiles; 4 waves
//    split the score tile (one 16-key column each), combine online-softmax
//    stats through LDS; P@V B-fragments read the L2-resident transposed
//    latent (latv[dim][key], key-contiguous) directly as global_load_b128
// ---------------------------------------------------------------------------

typedef __attribute__((ext_vector_type(16))) _Float16 v16h;
typedef __attribute__((ext_vector_type(8)))  float    v8f;
typedef __attribute__((ext_vector_type(4)))  int      v4i;

#define H_     128
#define D_     5120
#define QLORA  1536
#define KVL    512
#define ROPE_  64
#define NOPE_  128
#define VDIM_  128
#define QH     192            // NOPE + ROPE
#define KW     576            // KVL + ROPE
#define S_     1024
#define EPS_   1e-6f
#define SCALE_F 0.13523383f   // 192^-0.5 * (0.1*ln40+1)^2

#ifndef __has_builtin
#define __has_builtin(x) 0
#endif
#if __has_builtin(__builtin_amdgcn_global_load_async_to_lds_b128) && \
    __has_builtin(__builtin_amdgcn_s_wait_asynccnt)
#define USE_ASYNC_LDS 1
#else
#define USE_ASYNC_LDS 0
#endif

#if USE_ASYNC_LDS
__device__ __forceinline__ void async_b128(const _Float16* g, _Float16* l) {
  // Builtin signature (from compiler diagnostic): generic pointers to int4.
  __builtin_amdgcn_global_load_async_to_lds_b128((v4i*)g, (v4i*)l, 0, 0);
}
#endif

// -------------------------- WMMA fragment helpers --------------------------

__device__ __forceinline__ v8f wmma16(v16h a, v16h b, v8f c) {
  return __builtin_amdgcn_wmma_f32_16x16x32_f16(
      false, a, false, b, (short)0, c, false, false);
}

// A fragment 16x32 f16: lane&15 = row, lane half = +16 K stripe,
// 16 contiguous halves (same K order as frag_b).
__device__ __forceinline__ v16h frag_a(const _Float16* p, int ld) {
  int lane = threadIdx.x & 31;
  const _Float16* row = p + (lane & 15) * ld + ((lane >> 4) << 4);
  v16h r;
#pragma unroll
  for (int i = 0; i < 16; ++i) r[i] = row[i];
  return r;
}

// B fragment 32x16 f16 from a row-per-column tile Bt[n][k] (k contiguous):
// lane&15 = col n, lane half = +16 K stripe, 16 contiguous halves
// (-> 2x ds_load_b128 from LDS or 2x global_load_b128 from global).
__device__ __forceinline__ v16h frag_b(const _Float16* p, int ld) {
  int lane = threadIdx.x & 31;
  const _Float16* row = p + (lane & 15) * ld + ((lane >> 4) << 4);
  v16h r;
#pragma unroll
  for (int i = 0; i < 16; ++i) r[i] = row[i];
  return r;
}

// C/D 16x16 f32: vgpr v -> row v + 8*(lane>=16), col = lane&15.
template <typename CT>
__device__ __forceinline__ void store_c(CT* c, long ldc, v8f acc, float scale) {
  int lane = threadIdx.x & 31;
  int n  = lane & 15;
  int mb = (lane >> 4) << 3;
#pragma unroll
  for (int v = 0; v < 8; ++v) c[(long)(mb + v) * ldc + n] = (CT)(acc[v] * scale);
}

// ------------------------------ generic GEMM -------------------------------
// C[M,N] = cscale * (A[M,K] @ B[K,N]); A (f32 or f16), B f32, row-major;
// f16 WMMA compute, f32 accumulate. 128x128 block tile, 256 threads = 8
// waves, each wave 32x64 (8 WMMA/K-step). blockIdx.x indexes M (fastest) so
// a whole x-row of blocks reuses one 128-column B stripe out of L2.
// M % 128 == 0 at every call site; N guarded.

#define BM 128
#define BN 128
#define BK 32
#define BKP 40   // padded K stride (halves)

template <typename AT, typename CT>
__global__ __launch_bounds__(256)
void gemm_wmma(const AT* __restrict__ A, const float* __restrict__ B,
               CT* __restrict__ C, int M, int N, int K,
               long lda, long ldb, long ldc,
               long sA, long sB, long sC, float cscale) {
  __shared__ _Float16 sAs[BM][BKP];
  __shared__ _Float16 sBt[BN][BKP];
  int b = blockIdx.z;
  A += (long)b * sA; B += (long)b * sB; C += (long)b * sC;
  int m0 = blockIdx.x * BM;
  int n0 = blockIdx.y * BN;
  int tid  = threadIdx.x;
  int wave = tid >> 5;
  int wm = (wave >> 1) << 5;   // 0,32,64,96
  int wn = (wave & 1) << 6;    // 0,64
  v8f acc[2][4];
#pragma unroll
  for (int i = 0; i < 2; ++i)
#pragma unroll
    for (int j = 0; j < 4; ++j) acc[i][j] = {};

  for (int k0 = 0; k0 < K; k0 += BK) {
    for (int i = tid; i < BM * BK; i += 256) {   // stage A (row-major)
      int r = i >> 5, c = i & 31;
      sAs[r][c] = (_Float16)A[(long)(m0 + r) * lda + (k0 + c)];
    }
    for (int i = tid; i < BN * BK; i += 256) {   // stage B transposed Bt[n][k]
      int r = i >> 5, c = i & 31;
      sBt[r][c] = (n0 + r < N) ? (_Float16)B[(long)(k0 + c) * ldb + (n0 + r)]
                               : (_Float16)0.f;
    }
    if (k0 + BK < K) {                           // prefetch next tiles
      __builtin_prefetch(&A[(long)(m0 + (tid >> 1)) * lda + (k0 + BK) + ((tid & 1) << 4)], 0, 1);
      __builtin_prefetch(&B[(long)(k0 + BK + (tid >> 3)) * ldb + n0 + ((tid & 7) << 4)], 0, 1);
    }
    __syncthreads();
    v16h a0 = frag_a(&sAs[wm][0], BKP);
    v16h a1 = frag_a(&sAs[wm + 16][0], BKP);
    v16h bf[4];
#pragma unroll
    for (int j = 0; j < 4; ++j) bf[j] = frag_b(&sBt[wn + 16 * j][0], BKP);
#pragma unroll
    for (int j = 0; j < 4; ++j) {
      acc[0][j] = wmma16(a0, bf[j], acc[0][j]);
      acc[1][j] = wmma16(a1, bf[j], acc[1][j]);
    }
    __syncthreads();
  }
#pragma unroll
  for (int i = 0; i < 2; ++i)
#pragma unroll
    for (int j = 0; j < 4; ++j) {
      int col = n0 + wn + 16 * j;
      if (col < N)
        store_c(C + (long)(m0 + wm + 16 * i) * ldc + col, ldc, acc[i][j], cscale);
    }
}

// ------------------------------- RMSNorm (q_a) -----------------------------

__global__ __launch_bounds__(256)
void rmsnorm1536(const float* __restrict__ x, const float* __restrict__ w,
                 float* __restrict__ y) {
  __shared__ float red[256];
  int s = blockIdx.x;
  const float* row = x + (long)s * QLORA;
  float acc = 0.f;
  for (int i = threadIdx.x; i < QLORA; i += 256) { float v = row[i]; acc += v * v; }
  red[threadIdx.x] = acc;
  __syncthreads();
  for (int off = 128; off > 0; off >>= 1) {
    if ((int)threadIdx.x < off) red[threadIdx.x] += red[threadIdx.x + off];
    __syncthreads();
  }
  float inv = rsqrtf(red[0] / (float)QLORA + EPS_);
  for (int i = threadIdx.x; i < QLORA; i += 256)
    y[(long)s * QLORA + i] = row[i] * inv * w[i];
}

// -- kv post: latent RMSNorm + RoPE(k_pe) -> k_full (f16) + latv (V^T, f16) --

__global__ __launch_bounds__(512)
void kv_post(const float* __restrict__ kv, const float* __restrict__ w,
             const float* __restrict__ cosb, const float* __restrict__ sinb,
             _Float16* __restrict__ kfull, _Float16* __restrict__ latv) {
  __shared__ float red[512];
  int s = blockIdx.x;
  int t = threadIdx.x;
  const float* row = kv + (long)s * KW;
  float v = row[t];
  red[t] = v * v;
  __syncthreads();
  for (int off = 256; off > 0; off >>= 1) {
    if (t < off) red[t] += red[t + off];
    __syncthreads();
  }
  float inv = rsqrtf(red[0] / (float)KVL + EPS_);
  _Float16 lat = (_Float16)(v * inv * w[t]);
  kfull[(long)s * KW + t] = lat;
  latv[(long)t * S_ + s]  = lat;           // transposed copy (dim-major)
  if (t < ROPE_) {
    float x   = row[KVL + t];
    float rot = (t < 32) ? -row[KVL + t + 32] : row[KVL + t - 32];
    kfull[(long)s * KW + KVL + t] =
        (_Float16)(x * cosb[s * ROPE_ + t] + rot * sinb[s * ROPE_ + t]);
  }
}

// --- q RoPE: q[..,128:192] -> q_full[..,512:576] (f16, SCALE folded) -------

__global__ __launch_bounds__(64)
void q_rope(const float* __restrict__ q, const float* __restrict__ cosb,
            const float* __restrict__ sinb, _Float16* __restrict__ qfull) {
  int s = blockIdx.x, hh = blockIdx.y, j = threadIdx.x;
  const float* qp = q + (long)s * (H_ * QH) + (long)hh * QH + NOPE_;
  float x   = qp[j];
  float rot = (j < 32) ? -qp[j + 32] : qp[j - 32];
  qfull[(long)s * (H_ * KW) + (long)hh * KW + KVL + j] =
      (_Float16)((x * cosb[s * ROPE_ + j] + rot * sinb[s * ROPE_ + j]) * SCALE_F);
}

// ------------------------------ flash attention ----------------------------

#define KWP 584   // padded LDS stride (halves); row = 1168B (16B aligned)

__global__ __launch_bounds__(128)
void flash_attn(const _Float16* __restrict__ qfull,
                const _Float16* __restrict__ kfull,
                const _Float16* __restrict__ latv,
                _Float16* __restrict__ attn) {
  __shared__ _Float16 sQ[16][KWP];      // 18.3 KB
  __shared__ _Float16 sK[64][KWP];      // 73.0 KB
  __shared__ _Float16 sP[16][64];       //  2.0 KB
  __shared__ float    wred[2][4][16];   // [0]=row max, [1]=row sum per wave
  int q0 = blockIdx.x * 16;
  int hh = blockIdx.y;
  int tid = threadIdx.x, lane = tid & 31, wave = tid >> 5;
  int cb = wave * 128;                  // this wave's output-dim chunk
  int hlf = lane >> 4, nl = lane & 15;

  const _Float16* qbase = qfull + (long)q0 * (H_ * KW) + (long)hh * KW;
#if USE_ASYNC_LDS
  for (int i = tid; i < 16 * (KW / 8); i += 128) {
    int r = i / (KW / 8), c = (i - r * (KW / 8)) * 8;
    async_b128(qbase + (long)r * (H_ * KW) + c, &sQ[r][c]);
  }
  __builtin_amdgcn_s_wait_asynccnt(0);
#else
  for (int i = tid; i < 16 * KW; i += 128) {
    int r = i / KW, c = i - r * KW;
    sQ[r][c] = qbase[(long)r * (H_ * KW) + c];
  }
#endif

  v8f o[8];
  float mrow[8], lrow[8];
#pragma unroll
  for (int i = 0; i < 8; ++i) { o[i] = {}; mrow[i] = -3.0e38f; lrow[i] = 0.f; }

  int nkt = q0 / 64 + 1;                // causal: 64-key tiles up to q tile
  for (int kt = 0; kt < nkt; ++kt) {
    int kbase = kt * 64;
    __syncthreads();                    // prior sK/sP/wred readers done
    const _Float16* kb = kfull + (long)kbase * KW;
#if USE_ASYNC_LDS
    for (int i = tid; i < 64 * (KW / 8); i += 128) {
      int r = i / (KW / 8), c = (i - r * (KW / 8)) * 8;
      async_b128(kb + (long)r * KW + c, &sK[r][c]);
    }
    __builtin_amdgcn_s_wait_asynccnt(0);
#else
    for (int i = tid; i < 64 * KW; i += 128) {
      int r = i / KW, c = i - r * KW;
      sK[r][c] = kb[(long)r * KW + c];
    }
#endif
    __syncthreads();

    // ---- each wave: one 16-key column tile of the 16x64 score block ----
    v8f cs = {};
#pragma unroll
    for (int kc = 0; kc < KW / 32; ++kc) {   // 18 chained WMMAs per wave
      v16h aq = frag_a(&sQ[0][kc * 32], KWP);
      v16h bk = frag_b(&sK[16 * wave][kc * 32], KWP);
      cs = wmma16(aq, bk, cs);
    }

    int keyc = kbase + 16 * wave + nl;       // this lane's key column
    float tm[8];
#pragma unroll
    for (int v = 0; v < 8; ++v) {            // causal mask + local max
      int row = q0 + v + 8 * hlf;
      float s = (keyc <= row) ? cs[v] : -3.0e38f;
      cs[v] = s;
      tm[v] = s;
    }
#pragma unroll
    for (int sh = 1; sh < 16; sh <<= 1)
#pragma unroll
      for (int v = 0; v < 8; ++v)
        tm[v] = fmaxf(tm[v], __shfl_xor(tm[v], sh, 32));
    if (nl == 0) {
#pragma unroll
      for (int v = 0; v < 8; ++v) wred[0][wave][8 * hlf + v] = tm[v];
    }
    __syncthreads();

    float alpha[8];
#pragma unroll
    for (int v = 0; v < 8; ++v) {            // combine max across 4 waves
      int ri = 8 * hlf + v;
      float gm = fmaxf(fmaxf(wred[0][0][ri], wred[0][1][ri]),
                       fmaxf(wred[0][2][ri], wred[0][3][ri]));
      float mn = fmaxf(mrow[v], gm);
      alpha[v] = __expf(mrow[v] - mn);
      mrow[v]  = mn;
    }

    float rs[8];
#pragma unroll
    for (int v = 0; v < 8; ++v) {            // exp + local row sum
      float p = __expf(cs[v] - mrow[v]);
      cs[v] = p;
      rs[v] = p;
    }
#pragma unroll
    for (int sh = 1; sh < 16; sh <<= 1)
#pragma unroll
      for (int v = 0; v < 8; ++v) rs[v] += __shfl_xor(rs[v], sh, 32);
    if (nl == 0) {
#pragma unroll
      for (int v = 0; v < 8; ++v) wred[1][wave][8 * hlf + v] = rs[v];
    }
#pragma unroll
    for (int v = 0; v < 8; ++v)              // P tile -> LDS (A-frag layout)
      sP[v + 8 * hlf][16 * wave + nl] = (_Float16)cs[v];
    __syncthreads();

#pragma unroll
    for (int v = 0; v < 8; ++v) {            // combine sums across waves
      int ri = 8 * hlf + v;
      lrow[v] = lrow[v] * alpha[v] +
                (wred[1][0][ri] + wred[1][1][ri] +
                 wred[1][2][ri] + wred[1][3][ri]);
    }
#pragma unroll
    for (int t = 0; t < 8; ++t)              // rescale running O
#pragma unroll
      for (int v = 0; v < 8; ++v) o[t][v] *= alpha[v];

    // P (16x64) @ V-chunk: B fragments are key-contiguous rows of the
    // L2-resident transposed latent -> 2x global_load_b128 each, no LDS.
    const _Float16* vbase = latv + (long)cb * S_ + kbase;
    v16h ap0 = frag_a(&sP[0][0],  64);
    v16h ap1 = frag_a(&sP[0][32], 64);
#pragma unroll
    for (int t = 0; t < 8; ++t) {
      v16h bv0 = frag_b(vbase + (long)(t * 16) * S_,      S_);
      v16h bv1 = frag_b(vbase + (long)(t * 16) * S_ + 32, S_);
      o[t] = wmma16(ap0, bv0, o[t]);
      o[t] = wmma16(ap1, bv1, o[t]);
    }
  }

  float rl[8];
#pragma unroll
  for (int v = 0; v < 8; ++v) rl[v] = 1.f / lrow[v];
#pragma unroll
  for (int t = 0; t < 8; ++t) {
#pragma unroll
    for (int v = 0; v < 8; ++v) o[t][v] *= rl[v];
    store_c(&attn[(long)q0 * (H_ * KVL) + (long)hh * KVL + cb + t * 16],
            (long)H_ * KVL, o[t], 1.0f);
  }
}

// --------------------------------- launcher --------------------------------

extern "C" void kernel_launch(void* const* d_in, const int* in_sizes, int n_in,
                              void* d_out, int out_size, void* d_ws, size_t ws_size,
                              hipStream_t stream) {
  (void)in_sizes; (void)n_in; (void)out_size; (void)ws_size;
  const float* hid     = (const float*)d_in[0];
  const float* cosb    = (const float*)d_in[1];
  const float* sinb    = (const float*)d_in[2];
  const float* q_a_w   = (const float*)d_in[3];
  const float* q_a_ln  = (const float*)d_in[4];
  const float* q_b_w   = (const float*)d_in[5];
  const float* kv_a_w  = (const float*)d_in[6];
  const float* kv_a_ln = (const float*)d_in[7];
  const float* kc_w    = (const float*)d_in[8];
  const float* vc_w    = (const float*)d_in[9];
  const float* o_w     = (const float*)d_in[10];
  float* out = (float*)d_out;

  char* wsb = (char*)d_ws;
  size_t off = 0;
  auto alloc = [&](size_t bytes) {
    void* p = wsb + off;
    off = (off + bytes + 255) & ~(size_t)255;
    return p;
  };
  float*    qa    = (float*)   alloc((size_t)S_ * QLORA * 4);
  float*    qan   = (float*)   alloc((size_t)S_ * QLORA * 4);
  float*    q     = (float*)   alloc((size_t)S_ * H_ * QH * 4);
  float*    kv    = (float*)   alloc((size_t)S_ * KW * 4);
  _Float16* kfull = (_Float16*)alloc((size_t)S_ * KW * 2);
  _Float16* latv  = (_Float16*)alloc((size_t)KVL * S_ * 2);
  _Float16* qfull = (_Float16*)alloc((size_t)S_ * H_ * KW * 2);
  _Float16* attn  = (_Float16*)alloc((size_t)S_ * H_ * KVL * 2);
  float*    ctx   = (float*)   alloc((size_t)S_ * H_ * VDIM_ * 4);

  dim3 blk(256);
  // 1) qa = hid @ q_a_w                       (1024x1536, K=5120)
  gemm_wmma<float, float><<<dim3(S_ / 128, QLORA / 128, 1), blk, 0, stream>>>(
      hid, q_a_w, qa, S_, QLORA, D_, D_, QLORA, QLORA, 0, 0, 0, 1.0f);
  // 2) qan = rmsnorm(qa) * q_a_ln_w
  rmsnorm1536<<<S_, 256, 0, stream>>>(qa, q_a_ln, qan);
  // 3) q = qan @ q_b_w                        (1024x24576, K=1536)
  gemm_wmma<float, float><<<dim3(S_ / 128, (H_ * QH) / 128, 1), blk, 0, stream>>>(
      qan, q_b_w, q, S_, H_ * QH, QLORA, QLORA, H_ * QH, H_ * QH, 0, 0, 0, 1.0f);
  // 4) kv = hid @ kv_a_w                      (1024x576, K=5120; N guarded)
  gemm_wmma<float, float><<<dim3(S_ / 128, (KW + 127) / 128, 1), blk, 0, stream>>>(
      hid, kv_a_w, kv, S_, KW, D_, D_, KW, KW, 0, 0, 0, 1.0f);
  // 5) k_full = [rmsnorm(latent)*w, rope(k_pe)] + latv = latent^T   (f16)
  kv_post<<<S_, 512, 0, stream>>>(kv, kv_a_ln, cosb, sinb, kfull, latv);
  // 6) q_full[:, h, :512] = SCALE * q_nope_h @ kc_w[h]   (f16, batched on H)
  gemm_wmma<float, _Float16><<<dim3(S_ / 128, KVL / 128, H_), blk, 0, stream>>>(
      q, kc_w, qfull, S_, KVL, NOPE_,
      (long)H_ * QH, KVL, (long)H_ * KW,
      QH, (long)NOPE_ * KVL, KW, SCALE_F);
  // 7) q_full[:, h, 512:576] = SCALE * rope(q_pe)        (f16)
  q_rope<<<dim3(S_, H_), 64, 0, stream>>>(q, cosb, sinb, qfull);
  // 8) flash attention -> attn (S,H,512) f16
  flash_attn<<<dim3(S_ / 16, H_), 128, 0, stream>>>(qfull, kfull, latv, attn);
  // 9) ctx[:, h, :] = attn_h @ vc_w[h]        (batched on H, f16 A operand)
  gemm_wmma<_Float16, float><<<dim3(S_ / 128, VDIM_ / 128, H_), blk, 0, stream>>>(
      attn, vc_w, ctx, S_, VDIM_, KVL,
      (long)H_ * KVL, VDIM_, (long)H_ * VDIM_,
      KVL, (long)KVL * VDIM_, VDIM_, 1.0f);
  // 10) out = ctx @ o_w                       (1024x5120, K=16384)
  gemm_wmma<float, float><<<dim3(S_ / 128, D_ / 128, 1), blk, 0, stream>>>(
      ctx, o_w, out, S_, D_, H_ * VDIM_, (long)H_ * VDIM_, D_, D_, 0, 0, 0, 1.0f);
}
